// TrueSlidingWindowMHA_90142773609168
// MI455X (gfx1250) — compile-verified, matrix operand order checked by reading, not run on hardware
//
#include <hip/hip_runtime.h>
#include <hip/hip_bf16.h>
#include <stdint.h>
#include <stddef.h>

// ---------- problem constants ----------
#define B_   4
#define N_   4096
#define D_   1024
#define H_   16
#define DH_  64
#define W_   256
#define BH_  64          // B_*H_
#define TOK_ 16384       // B_*N_

typedef __attribute__((ext_vector_type(16))) _Float16 v16h;
typedef __attribute__((ext_vector_type(8)))  float    v8f;
typedef __attribute__((ext_vector_type(4)))  float    f4;
typedef __attribute__((ext_vector_type(4)))  int      i4;

// ---------- gfx1250 async global->LDS copy (ASYNCcnt path), with fallback ----
#if defined(__HIP_DEVICE_COMPILE__) && __has_builtin(__builtin_amdgcn_global_load_async_to_lds_b128)
#define HAVE_ASYNC_LDS 1
static __device__ __forceinline__ void async_cp16(const void* g, void* l) {
  __builtin_amdgcn_global_load_async_to_lds_b128(
      (__attribute__((address_space(1))) i4*)(g),
      (__attribute__((address_space(3))) i4*)(l),
      0, 0);
}
static __device__ __forceinline__ void async_wait0() {
#if __has_builtin(__builtin_amdgcn_s_wait_asynccnt)
  __builtin_amdgcn_s_wait_asynccnt(0);
#else
  asm volatile("s_wait_asynccnt 0x0" ::: "memory");
#endif
}
#else
#define HAVE_ASYNC_LDS 0
#endif

union Frag {
  v16h h;
  f4   q[2];
};

static __device__ __forceinline__ float rmax16(float v) {
#pragma unroll
  for (int m = 8; m >= 1; m >>= 1) v = fmaxf(v, __shfl_xor(v, m, 32));
  return v;  // xor masks < 16 keep the reduction inside each 16-lane half (wave32)
}
static __device__ __forceinline__ float rsum16(float v) {
#pragma unroll
  for (int m = 8; m >= 1; m >>= 1) v += __shfl_xor(v, m, 32);
  return v;
}

// =====================================================================
// Kernel 1: qkv = x @ w_qkv  (fp32 in, f16 WMMA, f32 accum)
// Epilogue scatters into head-major Q (scaled), K, and transposed V.
// Block tile 128x128, 8 waves (2x4), wave tile 64x32, K step 32.
// =====================================================================
__global__ __launch_bounds__(256) void qkv_gemm_kernel(
    const float* __restrict__ x, const float* __restrict__ wqkv,
    _Float16* __restrict__ qh, _Float16* __restrict__ kh,
    _Float16* __restrict__ vth)
{
  __shared__ _Float16 As[128][32];   // row-major A tile (M x K)
  __shared__ _Float16 Bs[128][32];   // transposed B tile (N x K)

  const int tid  = threadIdx.x;
  const int lane = tid & 31;
  const int wave = tid >> 5;
  const int wm   = wave >> 2;        // 0..1
  const int wn   = wave & 3;         // 0..3
  const int m0   = blockIdx.y * 128;
  const int n0   = blockIdx.x * 128;
  const int off  = (lane < 16) ? 0 : 1;   // half-wave K offset (units of 8 halves)
  const int l15  = lane & 15;
  const int hi8  = (lane < 16) ? 0 : 8;

  v8f acc[4][2];
#pragma unroll
  for (int mi = 0; mi < 4; ++mi)
#pragma unroll
    for (int ni = 0; ni < 2; ++ni)
#pragma unroll
      for (int e = 0; e < 8; ++e) acc[mi][ni][e] = 0.0f;

  for (int k0 = 0; k0 < D_; k0 += 32) {
#pragma unroll
    for (int it = 0; it < 4; ++it) {
      int fid = tid + it * 256;                 // 0..1023
      // A tile: 128 rows x 8 float4 (fp32 -> f16 convert)
      int r = fid >> 3, c4a = fid & 7;
      f4 av = *(const f4*)(x + (size_t)(m0 + r) * D_ + k0 + c4a * 4);
#pragma unroll
      for (int j = 0; j < 4; ++j) As[r][c4a * 4 + j] = (_Float16)av[j];
      // B tile: 32 k-rows x 32 float4, stored transposed
      int kk = fid >> 5, c4b = fid & 31;
      f4 bv = *(const f4*)(wqkv + (size_t)(k0 + kk) * (3 * D_) + n0 + c4b * 4);
#pragma unroll
      for (int j = 0; j < 4; ++j) Bs[c4b * 4 + j][kk] = (_Float16)bv[j];
    }
    __syncthreads();

    Frag af[4], bf[2];
#pragma unroll
    for (int mi = 0; mi < 4; ++mi) {
      const f4* p = (const f4*)&As[wm * 64 + mi * 16 + l15][0];
      af[mi].q[0] = p[off];          // K 0..7  (hi half: 8..15)
      af[mi].q[1] = p[off + 2];      // K 16..23 (hi half: 24..31)
    }
#pragma unroll
    for (int ni = 0; ni < 2; ++ni) {
      const f4* p = (const f4*)&Bs[wn * 32 + ni * 16 + l15][0];
      bf[ni].q[0] = p[off];
      bf[ni].q[1] = p[off + 2];
    }
#pragma unroll
    for (int mi = 0; mi < 4; ++mi)
#pragma unroll
      for (int ni = 0; ni < 2; ++ni)
        acc[mi][ni] = __builtin_amdgcn_wmma_f32_16x16x32_f16(
            false, af[mi].h, false, bf[ni].h, (short)0, acc[mi][ni], false, false);
    __syncthreads();
  }

  // Epilogue. The 128-wide N tile never crosses a 1024-aligned section
  // boundary, so the Q/K/V split is uniform per block -> one scalar branch.
  const int sec = n0 >> 10;          // 0=Q 1=K 2=V (uniform)
  if (sec < 2) {
    _Float16* __restrict__ dst = (sec == 0) ? qh : kh;
    const float scale = (sec == 0) ? 0.125f : 1.0f;   // DH^-0.5 for Q
#pragma unroll
    for (int mi = 0; mi < 4; ++mi)
#pragma unroll
      for (int ni = 0; ni < 2; ++ni) {
        int col = n0 + wn * 32 + ni * 16 + l15;
        int cc  = col & 1023;
        int hh  = cc >> 6;
        int dh  = cc & 63;
#pragma unroll
        for (int e = 0; e < 8; ++e) {
          int row = m0 + wm * 64 + mi * 16 + e + hi8;
          int b   = row >> 12;
          int tok = row & 4095;
          int bh  = b * H_ + hh;
          dst[((size_t)bh * N_ + tok) * DH_ + dh] =
              (_Float16)(acc[mi][ni][e] * scale);
        }
      }
  } else {
#pragma unroll
    for (int mi = 0; mi < 4; ++mi)
#pragma unroll
      for (int ni = 0; ni < 2; ++ni) {
        int col = n0 + wn * 32 + ni * 16 + l15;
        int cc  = col & 1023;
        int hh  = cc >> 6;
        int dh  = cc & 63;
#pragma unroll
        for (int e = 0; e < 8; ++e) {
          int row = m0 + wm * 64 + mi * 16 + e + hi8;
          int b   = row >> 12;
          int tok = row & 4095;
          int bh  = b * H_ + hh;
          vth[((size_t)bh * DH_ + dh) * N_ + tok] = (_Float16)acc[mi][ni][e];
        }
      }
  }
}

// =====================================================================
// Kernel 2: causal sliding-window flash attention.
// Grid: (64 query-blocks of 64, 64 batch*heads). 4 waves / block,
// each wave owns 16 query rows; keys streamed in chunks of 64
// (async global->LDS on the ASYNCcnt path when available).
// =====================================================================
__global__ __launch_bounds__(128) void swa_kernel(
    const _Float16* __restrict__ qh, const _Float16* __restrict__ kh,
    const _Float16* __restrict__ vth, _Float16* __restrict__ ao)
{
  __shared__ _Float16 Ks[64][64];     // [key][dh]
  __shared__ _Float16 Vts[64][64];    // [dh][key]  (V transposed)
  __shared__ _Float16 Ps[4][16][64];  // per-wave P staging [qrow][key]

  const int tid  = threadIdx.x;
  const int lane = tid & 31;
  const int w    = tid >> 5;          // wave 0..3
  const int l15  = lane & 15;
  const int off  = (lane < 16) ? 0 : 1;
  const int hi8  = (lane < 16) ? 0 : 8;
  const int blk  = blockIdx.x;        // query block (64 queries)
  const int bh   = blockIdx.y;
  const int qt0  = blk * 64;
  const int qrow = qt0 + w * 16 + l15;

  // Q A-fragments for K-dim dh 0..31 and 32..63 (loaded once).
  Frag qf[2];
  {
    const f4* p = (const f4*)(qh + ((size_t)bh * N_ + qrow) * DH_);
    qf[0].q[0] = p[off];     qf[0].q[1] = p[off + 2];
    qf[1].q[0] = p[off + 4]; qf[1].q[1] = p[off + 6];
  }

  float m_r[8], l_r[8];
  v8f Od[4];
#pragma unroll
  for (int e = 0; e < 8; ++e) { m_r[e] = -3.0e38f; l_r[e] = 0.0f; }
#pragma unroll
  for (int dt = 0; dt < 4; ++dt)
#pragma unroll
    for (int e = 0; e < 8; ++e) Od[dt][e] = 0.0f;

  const int c0 = (blk >= 4) ? (blk - 4) : 0;   // window spans <= 5 chunks of 64
  for (int c = c0; c <= blk; ++c) {
    const int kc0 = c * 64;
#pragma unroll
    for (int it = 0; it < 4; ++it) {
      int fid = tid + it * 128;                // 0..511
      int r = fid >> 3, c8 = fid & 7;
      const _Float16* gk = kh  + ((size_t)bh * N_ + kc0 + r) * DH_ + c8 * 8;
      const _Float16* gv = vth + ((size_t)bh * DH_ + r) * N_ + kc0 + c8 * 8;
#if HAVE_ASYNC_LDS
      async_cp16(gk, &Ks[r][c8 * 8]);
      async_cp16(gv, &Vts[r][c8 * 8]);
#else
      ((f4*)&Ks[r][0])[c8]  = *(const f4*)gk;
      ((f4*)&Vts[r][0])[c8] = *(const f4*)gv;
#endif
    }
    if (c < blk) {   // prefetch next chunk -> global_prefetch_b8
      __builtin_prefetch(kh  + ((size_t)bh * N_ + kc0 + 64 + (tid >> 1)) * DH_, 0, 1);
      __builtin_prefetch(vth + ((size_t)bh * DH_ + (tid >> 1)) * N_ + kc0 + 64, 0, 1);
    }
#if HAVE_ASYNC_LDS
    async_wait0();   // issuing wave's ASYNCcnt -> 0 before signaling barrier
#endif
    __syncthreads();

    // S = Q . K^T  (4 key sub-tiles of 16)
    v8f S[4];
#pragma unroll
    for (int nt = 0; nt < 4; ++nt) {
      Frag kb0, kb1;
      const f4* p = (const f4*)&Ks[nt * 16 + l15][0];
      kb0.q[0] = p[off];     kb0.q[1] = p[off + 2];   // dh 0..31
      kb1.q[0] = p[off + 4]; kb1.q[1] = p[off + 6];   // dh 32..63
      v8f cacc;
#pragma unroll
      for (int e = 0; e < 8; ++e) cacc[e] = 0.0f;
      cacc = __builtin_amdgcn_wmma_f32_16x16x32_f16(false, qf[0].h, false, kb0.h, (short)0, cacc, false, false);
      cacc = __builtin_amdgcn_wmma_f32_16x16x32_f16(false, qf[1].h, false, kb1.h, (short)0, cacc, false, false);
      S[nt] = cacc;
    }

    // Mask: key j valid iff i-W+1 <= j <= i (sentinel more negative than m_r init).
    float mx[8];
#pragma unroll
    for (int e = 0; e < 8; ++e) mx[e] = -3.0e38f;
#pragma unroll
    for (int nt = 0; nt < 4; ++nt) {
      int j = kc0 + nt * 16 + l15;
#pragma unroll
      for (int e = 0; e < 8; ++e) {
        int i = qt0 + w * 16 + e + hi8;
        bool valid = (j <= i) && (j >= i - (W_ - 1));
        float s = valid ? S[nt][e] : -3.402823e38f;
        S[nt][e] = s;
        mx[e] = fmaxf(mx[e], s);
      }
    }
#pragma unroll
    for (int e = 0; e < 8; ++e) mx[e] = rmax16(mx[e]);

    float alpha[8];
#pragma unroll
    for (int e = 0; e < 8; ++e) {
      float mn = fmaxf(m_r[e], mx[e]);
      alpha[e] = __expf(m_r[e] - mn);
      m_r[e] = mn;
    }
    float rs[8];
#pragma unroll
    for (int e = 0; e < 8; ++e) rs[e] = 0.0f;
#pragma unroll
    for (int nt = 0; nt < 4; ++nt)
#pragma unroll
      for (int e = 0; e < 8; ++e) {
        float pv = __expf(S[nt][e] - m_r[e]);   // masked -> exp(-huge) = 0
        S[nt][e] = pv;
        rs[e] += pv;
      }
#pragma unroll
    for (int e = 0; e < 8; ++e) {
      rs[e] = rsum16(rs[e]);
      l_r[e] = l_r[e] * alpha[e] + rs[e];
    }
#pragma unroll
    for (int dt = 0; dt < 4; ++dt)
#pragma unroll
      for (int e = 0; e < 8; ++e) Od[dt][e] *= alpha[e];

    // Stage P to LDS (C layout -> row-major) and reload in A layout.
    // Same-wave DS ops are in-order (DScnt); compiler inserts the waits.
#pragma unroll
    for (int nt = 0; nt < 4; ++nt)
#pragma unroll
      for (int e = 0; e < 8; ++e)
        Ps[w][e + hi8][nt * 16 + l15] = (_Float16)S[nt][e];

    Frag pa0, pa1;
    {
      const f4* p = (const f4*)&Ps[w][l15][0];
      pa0.q[0] = p[off];     pa0.q[1] = p[off + 2];   // keys 0..31
      pa1.q[0] = p[off + 4]; pa1.q[1] = p[off + 6];   // keys 32..63
    }
    // O += P . V   (B fragments from transposed-V tile: contiguous b128 loads)
#pragma unroll
    for (int dt = 0; dt < 4; ++dt) {
      Frag vb0, vb1;
      const f4* p = (const f4*)&Vts[dt * 16 + l15][0];
      vb0.q[0] = p[off];     vb0.q[1] = p[off + 2];
      vb1.q[0] = p[off + 4]; vb1.q[1] = p[off + 6];
      Od[dt] = __builtin_amdgcn_wmma_f32_16x16x32_f16(false, pa0.h, false, vb0.h, (short)0, Od[dt], false, false);
      Od[dt] = __builtin_amdgcn_wmma_f32_16x16x32_f16(false, pa1.h, false, vb1.h, (short)0, Od[dt], false, false);
    }
    __syncthreads();   // readers done before next chunk's (async) LDS writes
  }

  // Normalize and write head-interleaved f16: ao[b][tok][h*64+dh]
  const int b = bh >> 4, hh = bh & 15;
#pragma unroll
  for (int dt = 0; dt < 4; ++dt) {
    int dh = dt * 16 + l15;
#pragma unroll
    for (int e = 0; e < 8; ++e) {
      int tok = qt0 + w * 16 + e + hi8;
      float val = Od[dt][e] / l_r[e];
      ao[((size_t)b * N_ + tok) * (H_ * DH_) + hh * DH_ + dh] = (_Float16)val;
    }
  }
}

// =====================================================================
// Kernel 3: out = attn_out(f16) @ w_out(fp32->f16), f32 result.
// Same tiling as kernel 1; M=16384, N=1024, K=1024.
// A tile is already f16 -> async global->LDS when available.
// =====================================================================
__global__ __launch_bounds__(256) void out_gemm_kernel(
    const _Float16* __restrict__ ao, const float* __restrict__ wout,
    float* __restrict__ out)
{
  __shared__ _Float16 As[128][32];
  __shared__ _Float16 Bs[128][32];

  const int tid  = threadIdx.x;
  const int lane = tid & 31;
  const int wave = tid >> 5;
  const int wm   = wave >> 2;
  const int wn   = wave & 3;
  const int m0   = blockIdx.y * 128;
  const int n0   = blockIdx.x * 128;
  const int off  = (lane < 16) ? 0 : 1;
  const int l15  = lane & 15;
  const int hi8  = (lane < 16) ? 0 : 8;

  v8f acc[4][2];
#pragma unroll
  for (int mi = 0; mi < 4; ++mi)
#pragma unroll
    for (int ni = 0; ni < 2; ++ni)
#pragma unroll
      for (int e = 0; e < 8; ++e) acc[mi][ni][e] = 0.0f;

  for (int k0 = 0; k0 < D_; k0 += 32) {
#pragma unroll
    for (int it = 0; it < 2; ++it) {          // A already f16: straight b128 copies
      int fid = tid + it * 256;               // 0..511
      int r = fid >> 2, c8 = fid & 3;
      const _Float16* ga = ao + (size_t)(m0 + r) * D_ + k0 + c8 * 8;
#if HAVE_ASYNC_LDS
      async_cp16(ga, &As[r][c8 * 8]);
#else
      ((f4*)&As[r][0])[c8] = *(const f4*)ga;
#endif
    }
#pragma unroll
    for (int it = 0; it < 4; ++it) {          // B fp32 -> f16 transposed
      int fid = tid + it * 256;               // 0..1023
      int kk = fid >> 5, c4 = fid & 31;
      f4 bv = *(const f4*)(wout + (size_t)(k0 + kk) * D_ + n0 + c4 * 4);
#pragma unroll
      for (int j = 0; j < 4; ++j) Bs[c4 * 4 + j][kk] = (_Float16)bv[j];
    }
#if HAVE_ASYNC_LDS
    async_wait0();
#endif
    __syncthreads();

    Frag af[4], bf[2];
#pragma unroll
    for (int mi = 0; mi < 4; ++mi) {
      const f4* p = (const f4*)&As[wm * 64 + mi * 16 + l15][0];
      af[mi].q[0] = p[off];
      af[mi].q[1] = p[off + 2];
    }
#pragma unroll
    for (int ni = 0; ni < 2; ++ni) {
      const f4* p = (const f4*)&Bs[wn * 32 + ni * 16 + l15][0];
      bf[ni].q[0] = p[off];
      bf[ni].q[1] = p[off + 2];
    }
#pragma unroll
    for (int mi = 0; mi < 4; ++mi)
#pragma unroll
      for (int ni = 0; ni < 2; ++ni)
        acc[mi][ni] = __builtin_amdgcn_wmma_f32_16x16x32_f16(
            false, af[mi].h, false, bf[ni].h, (short)0, acc[mi][ni], false, false);
    __syncthreads();
  }

#pragma unroll
  for (int mi = 0; mi < 4; ++mi)
#pragma unroll
    for (int ni = 0; ni < 2; ++ni) {
      int col = n0 + wn * 32 + ni * 16 + l15;
#pragma unroll
      for (int e = 0; e < 8; ++e) {
        int row = m0 + wm * 64 + mi * 16 + e + hi8;
        out[(size_t)row * D_ + col] = acc[mi][ni][e];
      }
    }
}

// =====================================================================
extern "C" void kernel_launch(void* const* d_in, const int* in_sizes, int n_in,
                              void* d_out, int out_size, void* d_ws, size_t ws_size,
                              hipStream_t stream) {
  (void)in_sizes; (void)n_in; (void)out_size; (void)ws_size;
  const float* x    = (const float*)d_in[0];   // [B,N,D]
  const float* wqkv = (const float*)d_in[1];   // [D, 3*H*DH]
  const float* wout = (const float*)d_in[2];   // [H*DH, D]
  float* out = (float*)d_out;                  // [B,N,D]

  // workspace carve (f16): Q,K head-major [BH,N,DH]; V transposed [BH,DH,N];
  // attention output [B,N,H*DH].  4 x 32MB = 128MB.
  _Float16* qh  = (_Float16*)d_ws;
  _Float16* kh  = qh  + (size_t)BH_ * N_ * DH_;
  _Float16* vth = kh  + (size_t)BH_ * N_ * DH_;
  _Float16* ao  = vth + (size_t)BH_ * N_ * DH_;

  qkv_gemm_kernel<<<dim3(3 * D_ / 128, TOK_ / 128), 256, 0, stream>>>(x, wqkv, qh, kh, vth);
  swa_kernel<<<dim3(N_ / 64, BH_), 128, 0, stream>>>(qh, kh, vth, ao);
  out_gemm_kernel<<<dim3(D_ / 128, TOK_ / 128), 256, 0, stream>>>(ao, wout, out);
}
